// NNUE_6923487281305
// MI455X (gfx1250) — compile-verified, matrix operand order checked by reading, not run on hardware
//
#include <hip/hip_runtime.h>

typedef __attribute__((ext_vector_type(16))) __bf16 v16bf;
typedef __attribute__((ext_vector_type(8)))  float  v8f;

#define K1      40960
#define N1      256
#define KSPLIT  4
#define KSTEPS  (K1 / 32)          // 1280 k-steps of 32
#define KS_PER  (KSTEPS / KSPLIT)  // 320 per chunk
#define NT      16                 // 16 column tiles of 16

// one v_perm_b32: dst = { hi[31:16], lo[31:16] }  (truncated bf16 pair)
__device__ __forceinline__ unsigned pack_bf16x2(unsigned lo, unsigned hi) {
    return __builtin_amdgcn_perm(hi, lo, 0x07060302u);
}

union Frag { uint4 q[2]; unsigned u[8]; v16bf v; };
union AccU { v8f v; float f[8]; };

// Build a 16x32 bf16 A/B fragment (per ISA VGPR table) from 32 consecutive
// fp32 at rowPtr (row-major, this lane's row), one row per lane.
//   VGPR v, pair p: K = ((v>=4)?16:0) + (laneHi?8:0) + (v&3)*2 + p
__device__ __forceinline__ Frag make_frag_f32(const float* rowPtr, int hi) {
    const uint4* p = (const uint4*)(rowPtr + (hi << 3));
    uint4 q0 = p[0];          // K +0..3
    uint4 q1 = p[1];          // K +4..7
    uint4 q2 = p[4];          // K +16..19
    uint4 q3 = p[5];          // K +20..23
    Frag a;
    a.u[0] = pack_bf16x2(q0.x, q0.y);
    a.u[1] = pack_bf16x2(q0.z, q0.w);
    a.u[2] = pack_bf16x2(q1.x, q1.y);
    a.u[3] = pack_bf16x2(q1.z, q1.w);
    a.u[4] = pack_bf16x2(q2.x, q2.y);
    a.u[5] = pack_bf16x2(q2.z, q2.w);
    a.u[6] = pack_bf16x2(q3.x, q3.y);
    a.u[7] = pack_bf16x2(q3.z, q3.w);
    return a;
}

// ---------------------------------------------------------------------------
// Kernel 0: W1 (256 x 40960 fp32, row-major) -> bf16 fragments.
// Fragment f = nt*KSTEPS + ks holds the 16x32 B-tile (cols nt*16.., K ks*32..)
// laid out exactly as the WMMA B operand: 32 lanes x 8 dwords, contiguous.
// ---------------------------------------------------------------------------
__global__ __launch_bounds__(256) void convert_w1(const float* __restrict__ W1,
                                                  unsigned* __restrict__ W1bf)
{
    const int g    = blockIdx.x * 256 + threadIdx.x;  // frag*32 + lane
    const int lane = g & 31;
    const int f    = g >> 5;
    const int ks   = f % KSTEPS;
    const int nt   = f / KSTEPS;
    const int l16  = lane & 15;
    const int hi   = lane >> 4;

    const float* src = W1 + (size_t)(nt * 16 + l16) * K1 + ks * 32;
    Frag b = make_frag_f32(src, hi);
    unsigned* dst = W1bf + ((size_t)f * 32 + lane) * 8;
    *(uint4*)(dst)     = b.q[0];
    *(uint4*)(dst + 4) = b.q[1];
}

// ---------------------------------------------------------------------------
// Kernel 1: partial GEMM.  Xr[4096,40960] @ W1.T -> P[kc][4096,256]
// Block: 64 rows x 256 cols, one K chunk. 8 waves = 2 Mw x 4 Nw;
// wave = 32 rows x 64 cols = 8 accumulators.
// ---------------------------------------------------------------------------
__global__ __launch_bounds__(256) void gemm_partial(const float* __restrict__ x,
                                                    const unsigned* __restrict__ W1bf,
                                                    float* __restrict__ P,
                                                    int rbCount, int rows)
{
    const int rb   = blockIdx.x % rbCount;
    const int kc   = blockIdx.x / rbCount;
    const int tid  = threadIdx.x;
    const int wave = tid >> 5;
    const int lane = tid & 31;
    const int mw   = wave >> 2;       // 0..1  (32-row slab)
    const int nw   = wave & 3;        // 0..3  (64-col slab)
    const int l16  = lane & 15;
    const int hi   = lane >> 4;

    const int rowBase = rb * 64 + mw * 32;

    v8f acc[2][4];
    const v8f vzero = {0.f,0.f,0.f,0.f,0.f,0.f,0.f,0.f};
    #pragma unroll
    for (int a = 0; a < 2; ++a)
        #pragma unroll
        for (int t = 0; t < 4; ++t) acc[a][t] = vzero;

    const float* aRow0 = x + (size_t)(rowBase + l16) * K1;
    const float* aRow1 = x + (size_t)(rowBase + 16 + l16) * K1;
    // B fragments for this wave's 4 tiles, this K chunk, contiguous per tile
    const unsigned* bBase = W1bf + ((size_t)(nw * 4) * KSTEPS + kc * KS_PER) * 32 * 8;

    const int ksEnd = (kc + 1) * KS_PER;
    for (int ks = kc * KS_PER; ks < ksEnd; ++ks) {
        const int k = ks * 32;
        __builtin_prefetch(aRow0 + k + 32, 0, 0);
        __builtin_prefetch(aRow1 + k + 32, 0, 0);

        Frag a0 = make_frag_f32(aRow0 + k, hi);
        Frag a1 = make_frag_f32(aRow1 + k, hi);

        #pragma unroll
        for (int t = 0; t < 4; ++t) {
            const unsigned* bp =
                bBase + (((size_t)t * KSTEPS + (ks - kc * KS_PER)) * 32 + lane) * 8;
            Frag b;
            b.q[0] = *(const uint4*)(bp);
            b.q[1] = *(const uint4*)(bp + 4);
            acc[0][t] = __builtin_amdgcn_wmma_f32_16x16x32_bf16(
                false, a0.v, false, b.v, (short)0, acc[0][t], false, false);
            acc[1][t] = __builtin_amdgcn_wmma_f32_16x16x32_bf16(
                false, a1.v, false, b.v, (short)0, acc[1][t], false, false);
        }
    }

    // C/D layout: VGPR e -> M = e + laneHi*8, N = lane&15
    float* Pb = P + (size_t)kc * rows * N1;
    #pragma unroll
    for (int a = 0; a < 2; ++a) {
        #pragma unroll
        for (int t = 0; t < 4; ++t) {
            AccU c; c.v = acc[a][t];
            const int col = nw * 64 + t * 16 + l16;
            #pragma unroll
            for (int e = 0; e < 8; ++e) {
                const int row = rowBase + a * 16 + hi * 8 + e;
                Pb[(size_t)row * N1 + col] = c.f[e];
            }
        }
    }
}

// ---------------------------------------------------------------------------
// Kernel 2: reduce K-split partials + layer_norm(256) + leaky -> H[4096,256]
// ---------------------------------------------------------------------------
__global__ __launch_bounds__(256) void reduce_ln(const float* __restrict__ P,
                                                 float* __restrict__ H, int rows)
{
    __shared__ float s[256];
    __shared__ float q[256];
    const int row = blockIdx.x;
    const int t   = threadIdx.x;

    float v = 0.f;
    #pragma unroll
    for (int c = 0; c < KSPLIT; ++c)
        v += P[(size_t)c * rows * N1 + (size_t)row * N1 + t];

    s[t] = v; q[t] = v * v;
    __syncthreads();
    #pragma unroll
    for (int off = 128; off > 0; off >>= 1) {
        if (t < off) { s[t] += s[t + off]; q[t] += q[t + off]; }
        __syncthreads();
    }
    const float mu = s[0] * (1.0f / 256.0f);
    const float rs = rsqrtf(q[0] * (1.0f / 256.0f) - mu * mu);
    const float y  = (v - mu) * rs;
    H[(size_t)row * N1 + t] = fmaxf(0.05f * y, y);
}

// ---------------------------------------------------------------------------
// Kernel 3: tail layers (tiny FLOPs). One block per batch element.
// ---------------------------------------------------------------------------
__global__ __launch_bounds__(128) void tail_kernel(const float* __restrict__ H,
                                                   const float* __restrict__ W2,
                                                   const float* __restrict__ W3,
                                                   const float* __restrict__ W4,
                                                   float* __restrict__ out)
{
    __shared__ float hin[512];
    __shared__ float tv[128];
    __shared__ float hcat[128];
    __shared__ float uvec[8];
    __shared__ float stats[4];

    const int b   = blockIdx.x;
    const int tid = threadIdx.x;

    const float* src = H + (size_t)(2 * b) * 256;
    #pragma unroll
    for (int i = 0; i < 4; ++i) hin[tid + i * 128] = src[tid + i * 128];
    __syncthreads();

    {   // 256 -> 64 for each half
        const int half = tid >> 6;
        const int j    = tid & 63;
        const float* hv = hin + half * 256;
        const float* wr = W2 + j * 256;
        float sacc = 0.f;
        for (int kk = 0; kk < 256; ++kk) sacc += hv[kk] * wr[kk];
        tv[tid] = sacc;
    }
    __syncthreads();
    if (tid == 0 || tid == 64) {
        float S = 0.f, Q = 0.f;
        for (int i = 0; i < 64; ++i) { float v = tv[tid + i]; S += v; Q += v * v; }
        const float mu  = S * (1.0f / 64.0f);
        stats[(tid >> 6) * 2 + 0] = mu;
        stats[(tid >> 6) * 2 + 1] = rsqrtf(Q * (1.0f / 64.0f) - mu * mu);
    }
    __syncthreads();
    {
        const int half = tid >> 6;
        float v = (tv[tid] - stats[half * 2]) * stats[half * 2 + 1];
        hcat[tid] = fmaxf(0.05f * v, v);
    }
    __syncthreads();
    if (tid < 8) {
        const float* wr = W3 + tid * 128;
        float sacc = 0.f;
        for (int kk = 0; kk < 128; ++kk) sacc += hcat[kk] * wr[kk];
        uvec[tid] = sacc;
    }
    __syncthreads();
    if (tid == 0) {
        float S = 0.f, Q = 0.f;
        #pragma unroll
        for (int i = 0; i < 8; ++i) { S += uvec[i]; Q += uvec[i] * uvec[i]; }
        const float mu = S * 0.125f;
        const float rs = rsqrtf(Q * 0.125f - mu * mu);
        float o = 0.f;
        #pragma unroll
        for (int i = 0; i < 8; ++i) {
            float v = (uvec[i] - mu) * rs;
            o += fmaxf(0.05f * v, v) * W4[i];
        }
        out[b] = o;
    }
}

extern "C" void kernel_launch(void* const* d_in, const int* in_sizes, int n_in,
                              void* d_out, int out_size, void* d_ws, size_t ws_size,
                              hipStream_t stream)
{
    const float* x  = (const float*)d_in[0];
    const float* W1 = (const float*)d_in[1];
    const float* W2 = (const float*)d_in[2];
    const float* W3 = (const float*)d_in[3];
    const float* W4 = (const float*)d_in[4];
    float* out = (float*)d_out;

    const int rows    = in_sizes[0] / K1;   // 4096 (= 2 * batch)
    const int rbCount = rows / 64;          // 64 row-blocks

    // workspace layout
    float*    H    = (float*)d_ws;                                   //  4 MB
    float*    P    = (float*)((char*)d_ws + (size_t)rows * N1 * 4);  // 16 MB
    unsigned* W1bf = (unsigned*)((char*)d_ws +
                     (size_t)rows * N1 * 4 * (1 + KSPLIT));          // 21 MB

    convert_w1  <<<(NT * KSTEPS * 32) / 256, 256, 0, stream>>>(W1, W1bf);
    gemm_partial<<<rbCount * KSPLIT,         256, 0, stream>>>(x, W1bf, P, rbCount, rows);
    reduce_ln   <<<rows,                     256, 0, stream>>>(P, H, rows);
    tail_kernel <<<rows / 2,                 128, 0, stream>>>(H, W2, W3, W4, out);
}
